// ProtoLayer_10350871183810
// MI455X (gfx1250) — 8-bit matrix operand order repaired and checked by reading, compiled, not run on hardware
//
#include <hip/hip_runtime.h>

typedef __attribute__((ext_vector_type(8))) int   v8i;
typedef __attribute__((ext_vector_type(4))) float v4f;

namespace {
constexpr int BB  = 4;
constexpr int Q   = 75;
constexpr int P   = 5;
constexpr int D   = 640;
constexpr int NPAIRS = D * (D - 1) / 2;
constexpr int STR = 644;
constexpr int QT  = 5;
constexpr int NW  = 16;
constexpr int TPB = NW * 32;
constexpr int KSPLIT = 32;
constexpr int OUTN = BB * Q * P;

constexpr int count_groups() {
    int n = 0;
    for (int i = 0; i < D - 1; ++i) n += (D / 4) - ((i + 1) >> 2);
    return n;
}
constexpr int NGROUPS = count_groups();
static_assert(NGROUPS % 16 == 0, "chunks must be whole");
constexpr int NCHUNK = NGROUPS / 16;

constexpr int SQ_ELEMS = (QT * 16) * STR;
constexpr int SP_ELEMS = P * STR;
constexpr size_t SMEM_BYTES =
    (size_t)(SQ_ELEMS + SP_ELEMS) * sizeof(float) + (size_t)QT * 256 * sizeof(int);
}

__global__ void build_table_kernel(unsigned* __restrict__ tab) {
    int i = blockIdx.x * blockDim.x + threadIdx.x;
    if (i >= D - 1) return;
    int q = i >> 2, r = i & 3;
    int g0  = (D / 4) * i - (2 * q * (q - 1) + q * (r + 1));
    int jg0 = (i + 1) >> 2;
    int ng  = (D / 4) - jg0;
    unsigned base = (unsigned)i << 16;
    for (int t = 0; t < ng; ++t)
        tab[g0 + t] = base | (unsigned)(4 * (jg0 + t));
}

__global__ void zero_kernel(int* __restrict__ acc, int n) {
    int i = blockIdx.x * blockDim.x + threadIdx.x;
    if (i < n) acc[i] = 0;
}

__global__ void finalize_kernel(const int* __restrict__ acc, float* __restrict__ out, int n) {
    int i = blockIdx.x * blockDim.x + threadIdx.x;
    if (i < n) out[i] = (float)acc[i] / (float)NPAIRS;
}

__device__ __forceinline__ unsigned pack_groupA(const float* __restrict__ row, unsigned e) {
    const int i  = (int)(e >> 16);
    const int j0 = (int)(e & 0xFFFFu);
    const v4f x  = *(const v4f*)(row + j0);
    const float xi = row[i];
    unsigned dw = 0u;
#pragma unroll
    for (int t = 0; t < 4; ++t) {
        float d = x[t] - xi;
        unsigned by = (d > 0.f ? 0x01u : 0u) | (d < 0.f ? 0xFFu : 0u);
        dw |= by << (8 * t);
    }
    return dw;
}

__device__ __forceinline__ unsigned pack_groupB(const float* __restrict__ row, unsigned e) {
    const int i  = (int)(e >> 16);
    const int j0 = (int)(e & 0xFFFFu);
    const v4f x  = *(const v4f*)(row + j0);
    const float xi = row[i];
    unsigned dw = 0u;
#pragma unroll
    for (int t = 0; t < 4; ++t) {
        float d = x[t] - xi;
        unsigned by = (d > 0.f ? 0x01u : 0u) | (d < 0.f ? 0xFFu : 0u);
        by = (j0 + t > i) ? by : 0u;
        dw |= by << (8 * t);
    }
    return dw;
}

__global__ __launch_bounds__(TPB) void kendall_wmma_kernel(
    const float* __restrict__ query,
    const float* __restrict__ proto,
    const unsigned* __restrict__ tab,
    int* __restrict__ accum)
{
    extern __shared__ char smem[];
    float* sQ   = (float*)smem;
    float* sP   = sQ + SQ_ELEMS;
    int*   sAcc = (int*)(sP + SP_ELEMS);

    const int b   = blockIdx.x;
    const int zz  = blockIdx.y;
    const int tid = threadIdx.x;

    for (int idx = tid; idx < QT * 16 * D; idx += TPB) {
        int r = idx / D, c = idx - r * D;
        sQ[r * STR + c] = (r < Q) ? query[((size_t)b * Q + r) * D + c] : 0.f;
    }
    for (int idx = tid; idx < P * D; idx += TPB) {
        int r = idx / D, c = idx - r * D;
        sP[r * STR + c] = proto[((size_t)b * P + r) * D + c];
    }
    for (int idx = tid; idx < QT * 256; idx += TPB) sAcc[idx] = 0;
    __syncthreads();

    const int wave = tid >> 5;
    const int lane = tid & 31;
    const int m    = lane & 15;
    const int h    = lane >> 4;
    const bool valN = (m < P);
    const float* prow = sP + (valN ? m : 0) * STR;

    v8i accs[QT];
#pragma unroll
    for (int qt = 0; qt < QT; ++qt) accs[qt] = (v8i){0, 0, 0, 0, 0, 0, 0, 0};

    for (int kc = zz * NW + wave; kc < NCHUNK; kc += KSPLIT * NW) {
        const unsigned* te = tab + (size_t)kc * 16;

        unsigned eA[8], eB[8];
#pragma unroll
        for (int v = 0; v < 8; ++v) {
            eA[v] = te[((v >> 2) << 3) + (h << 2) + (v & 3)];
            eB[v] = te[((v >> 2) << 3) + (h << 2) + (v & 3)];
        }

        v8i bm;
#pragma unroll
        for (int v = 0; v < 8; ++v)
            bm[v] = valN ? (int)pack_groupB(prow, eB[v]) : 0;

#pragma unroll
        for (int qt = 0; qt < QT; ++qt) {
            const float* qrow = sQ + (qt * 16 + m) * STR;
            v8i a;
#pragma unroll
            for (int v = 0; v < 8; ++v)
                a[v] = (int)pack_groupA(qrow, eA[v]);
            accs[qt] = __builtin_amdgcn_wmma_i32_16x16x64_iu8(
                 true, a,  true, bm, accs[qt],
                 false,  false);
        }
    }

#pragma unroll
    for (int qt = 0; qt < QT; ++qt) {
#pragma unroll
        for (int v = 0; v < 8; ++v) {
            int M = v + (h << 3);
            atomicAdd(&sAcc[qt * 256 + (M << 4) + m], accs[qt][v]);
        }
    }
    __syncthreads();

    for (int idx = tid; idx < QT * 256; idx += TPB) {
        int qt = idx >> 8, rem = idx & 255;
        int M = rem >> 4, N = rem & 15;
        int q = qt * 16 + M;
        if (q < Q && N < P)
            atomicAdd(&accum[((size_t)b * Q + q) * P + N], sAcc[idx]);
    }
}

extern "C" void kernel_launch(void* const* d_in, const int* in_sizes, int n_in,
                              void* d_out, int out_size, void* d_ws, size_t ws_size,
                              hipStream_t stream) {
    (void)in_sizes; (void)n_in; (void)out_size; (void)ws_size;
    const float* query = (const float*)d_in[0];
    const float* proto = (const float*)d_in[1];
    float* out = (float*)d_out;

    int*      acc = (int*)d_ws;
    unsigned* tab = (unsigned*)((char*)d_ws + 8192);

    zero_kernel<<<dim3((OUTN + 255) / 256), 256, 0, stream>>>(acc, OUTN);
    build_table_kernel<<<dim3((D - 1 + 255) / 256), 256, 0, stream>>>(tab);
    kendall_wmma_kernel<<<dim3(BB, KSPLIT), TPB, SMEM_BYTES, stream>>>(query, proto, tab, acc);
    finalize_kernel<<<dim3((OUTN + 255) / 256), 256, 0, stream>>>(acc, out, OUTN);
}
